// ConnectFourPolicy_40776419508487
// MI455X (gfx1250) — matrix operand order repaired and checked by reading, not compiled, not run on hardware
//
#include <hip/hip_runtime.h>

typedef __attribute__((ext_vector_type(16))) _Float16 v16h;
typedef __attribute__((ext_vector_type(8)))  float    v8f;
typedef __attribute__((ext_vector_type(4)))  unsigned int u32x4;
typedef __attribute__((ext_vector_type(4)))  float    f32x4;

// ---------------- constants ----------------
#define BN_TOTAL 65536
#define DM   64
#define FF   128
#define NL   14
#define BOARDN 42
#define ACT  7

// f16 weight workspace layout (offsets in halves)
#define OFF_WIN  0                       // [64][64]  (K padded 50->64 with zeros)
#define OFF_WV   4096                    // NL x [64][64]
#define OFF_WO   (OFF_WV  + NL*4096)     // NL x [64][64]
#define OFF_W1   (OFF_WO  + NL*4096)     // NL x [128][64]
#define OFF_W2   (OFF_W1  + NL*8192)     // NL x [64][128]
#define OFF_WF   (OFF_W2  + NL*8192)     // [128][64]
#define OFF_WP1  (OFF_WF  + 8192)        // [128][128]
#define OFF_WP2  (OFF_WP1 + 16384)       // [128][128]
#define OFF_WA   (OFF_WP2 + 16384)       // [16][128] (rows 7..15 zero)

// LDS row pitches (halves / floats)
#define PH 72     // f16 activation buffer pitch (64 cols used)
#define PS 136    // f16 wide buffer pitch (up to 128 cols)
#define PX 68     // f32 scratch pitch (64 cols used)

union F16Frag { v16h v; u32x4 q[2]; };

// ---------------- weight conversion kernels (f32 -> f16 in ws) ----------------
__global__ void k_cvt_plain(const float* __restrict__ src, _Float16* __restrict__ dst, int n) {
    for (int i = blockIdx.x * blockDim.x + threadIdx.x; i < n; i += gridDim.x * blockDim.x)
        dst[i] = (_Float16)src[i];
}
// extract V block (rows 128..191) of each layer's Wqkv [L,192,64]
__global__ void k_cvt_wv(const float* __restrict__ Wqkv, _Float16* __restrict__ dst) {
    int n = NL * 4096;
    for (int i = blockIdx.x * blockDim.x + threadIdx.x; i < n; i += gridDim.x * blockDim.x) {
        int l = i >> 12, rem = i & 4095;
        dst[i] = (_Float16)Wqkv[l * 12288 + 8192 + rem];
    }
}
// W_in [64,50] -> [64,64] zero-padded in K
__global__ void k_cvt_win(const float* __restrict__ Win, _Float16* __restrict__ dst) {
    for (int i = blockIdx.x * blockDim.x + threadIdx.x; i < 4096; i += gridDim.x * blockDim.x) {
        int nn = i >> 6, k = i & 63;
        dst[i] = (_Float16)(k < 50 ? Win[nn * 50 + k] : 0.f);
    }
}
// Wa [7,128] -> [16,128] zero-padded in N
__global__ void k_cvt_wa(const float* __restrict__ Wa, _Float16* __restrict__ dst) {
    for (int i = blockIdx.x * blockDim.x + threadIdx.x; i < 2048; i += gridDim.x * blockDim.x) {
        int nn = i >> 7, k = i & 127;
        dst[i] = (_Float16)(nn < ACT ? Wa[nn * 128 + k] : 0.f);
    }
}

// ---------------- WMMA GEMM core ----------------
// D(16 x NT*16) = A(16 x KCH*32) @ B^T, A from LDS (row-major f16, pitch apitch),
// W row-major [N][K] f16 in global (K = KCH*32).
// A frag layout (ISA 16-bit A 16x32): lane(M=l&15, hi=l>>4): halves j<8 -> K=32c+8*hi+j,
//                                     j>=8 -> K=32c+16+8*hi+(j-8).
// B frag layout (same as A): lane(N=l&15, hi): halves j<8 -> K=32c+8*hi+j, j>=8 -> K=32c+16+8*hi+(j-8).
template <int KCH, int NT>
__device__ __forceinline__ void wmma_gemm(const _Float16* __restrict__ A, int apitch,
                                          const _Float16* __restrict__ W, int kdim,
                                          int lane, v8f* acc) {
    const int r16 = lane & 15, hi = lane >> 4;
    F16Frag a[KCH];
    const _Float16* ap = A + r16 * apitch + 8 * hi;
#pragma unroll
    for (int c = 0; c < KCH; ++c) {
        a[c].q[0] = *(const u32x4*)(ap + 32 * c);
        a[c].q[1] = *(const u32x4*)(ap + 32 * c + 16);
    }
#pragma unroll
    for (int t = 0; t < NT; ++t) {
        v8f c = {};
        const _Float16* wp = W + (t * 16 + r16) * kdim + 8 * hi;
#pragma unroll
        for (int cc = 0; cc < KCH; ++cc) {
            F16Frag b;
            b.q[0] = *(const u32x4*)(wp + 32 * cc);
            b.q[1] = *(const u32x4*)(wp + 32 * cc + 16);
            c = __builtin_amdgcn_wmma_f32_16x16x32_f16(false, a[cc].v, false, b.v,
                                                       (short)0, c, false, false);
        }
        acc[t] = c;
    }
}

// store D tiles (+bias, opt. relu) as f16 into LDS row-major buffer
template <int NT, bool RELU>
__device__ __forceinline__ void store_h16(const v8f* acc, _Float16* dst, int pitch,
                                          const float* __restrict__ bias, int lane) {
    const int n16 = lane & 15, hi = lane >> 4;
#pragma unroll
    for (int t = 0; t < NT; ++t) {
        float bn = bias[t * 16 + n16];
#pragma unroll
        for (int r = 0; r < 8; ++r) {
            float v = acc[t][r] + bn;
            if (RELU) v = v > 0.f ? v : 0.f;
            dst[(r + 8 * hi) * pitch + t * 16 + n16] = (_Float16)v;
        }
    }
}
// store D tiles (+bias) as f32 into LDS scratch (pitch PX)
template <int NT>
__device__ __forceinline__ void store_f32(const v8f* acc, float* dst,
                                          const float* __restrict__ bias, int lane) {
    const int n16 = lane & 15, hi = lane >> 4;
#pragma unroll
    for (int t = 0; t < NT; ++t) {
        float bn = bias[t * 16 + n16];
#pragma unroll
        for (int r = 0; r < 8; ++r)
            dst[(r + 8 * hi) * PX + t * 16 + n16] = acc[t][r] + bn;
    }
}

// LayerNorm( hres + x32 ) -> hres (f32) and hf (f16). lane = (row, 32-col half).
__device__ __forceinline__ void ln_pass(float* hres, const float* x32, _Float16* hf,
                                        const float* __restrict__ g,
                                        const float* __restrict__ b, int lane) {
    const int row = lane & 15, hb = (lane >> 4) * 32;
    float vals[32];
    float s = 0.f, s2 = 0.f;
    const f32x4* hp = (const f32x4*)(hres + row * PX + hb);
    const f32x4* xp = (const f32x4*)(x32 + row * PX + hb);
#pragma unroll
    for (int q = 0; q < 8; ++q) {
        f32x4 hv = hp[q], xv = xp[q];
#pragma unroll
        for (int e = 0; e < 4; ++e) {
            float t = hv[e] + xv[e];
            vals[q * 4 + e] = t;
            s += t; s2 += t * t;
        }
    }
    s  += __shfl_xor(s, 16, 32);
    s2 += __shfl_xor(s2, 16, 32);
    float mu  = s * (1.f / 64.f);
    float var = s2 * (1.f / 64.f) - mu * mu;
    float rs  = rsqrtf(var + 1e-5f);
    const f32x4* gp = (const f32x4*)(g + hb);
    const f32x4* bp = (const f32x4*)(b + hb);
    f32x4* ho = (f32x4*)(hres + row * PX + hb);
#pragma unroll
    for (int q = 0; q < 8; ++q) {
        f32x4 gv = gp[q], bv = bp[q], y;
#pragma unroll
        for (int e = 0; e < 4; ++e) y[e] = (vals[q * 4 + e] - mu) * rs * gv[e] + bv[e];
        ho[q] = y;
#pragma unroll
        for (int e = 0; e < 4; ++e) hf[row * PH + hb + q * 4 + e] = (_Float16)y[e];
    }
}

// ---------------- main fused kernel: 4 waves/block, 16 batch rows per wave ----------------
__launch_bounds__(128)
__global__ void k_policy_forward(
    const float* __restrict__ board, const int* __restrict__ mark,
    const float* __restrict__ emb,
    const float* __restrict__ b_in,  const float* __restrict__ bqkv,
    const float* __restrict__ bo,    const float* __restrict__ ln1w,
    const float* __restrict__ ln1b,  const float* __restrict__ b1,
    const float* __restrict__ b2,    const float* __restrict__ ln2w,
    const float* __restrict__ ln2b,  const float* __restrict__ bf,
    const float* __restrict__ bp1,   const float* __restrict__ bp2,
    const float* __restrict__ ba,
    const _Float16* __restrict__ ws, float* __restrict__ out) {
    __shared__ __align__(16) _Float16 sh_hf[4][16 * PH];   // f16 activations (A operand)
    __shared__ __align__(16) _Float16 sh_sf[4][16 * PS];   // f16 wide buffer (A operand)
    __shared__ __align__(16) float    sh_x[4][16 * PX];    // f32 GEMM-out scratch
    __shared__ __align__(16) float    sh_h[4][16 * PX];    // f32 residual stream

    const int wave = threadIdx.x >> 5, lane = threadIdx.x & 31;
    const int R0 = blockIdx.x * 64 + wave * 16;
    _Float16* hf = sh_hf[wave];
    _Float16* sf = sh_sf[wave];
    float*    x32  = sh_x[wave];
    float*    hres = sh_h[wave];

    const int row = lane & 15, hi = lane >> 4;
    const int grow = R0 + row;

    // --- stage input x50 = [board | mark_emb], zero-padded to K=64, as f16 ---
    {
        const int mi = mark[grow] - 1;   // {1,2} -> {0,1}
        #pragma unroll
        for (int i = 0; i < 32; ++i) {
            int k = hi * 32 + i;
            float v = (k < BOARDN) ? board[grow * BOARDN + k]
                     : (k < 50)    ? emb[mi * 8 + (k - BOARDN)]
                                   : 0.f;
            hf[row * PH + k] = (_Float16)v;
        }
    }

    v8f acc4[4];
    v8f acc8[8];

    // --- input projection: [16x64] = x50pad @ Win^T + b_in ---
    wmma_gemm<2, 4>(hf, PH, ws + OFF_WIN, DM, lane, acc4);
    {
        const int n16 = lane & 15;
        #pragma unroll
        for (int t = 0; t < 4; ++t) {
            float bn = b_in[t * 16 + n16];
            #pragma unroll
            for (int r = 0; r < 8; ++r) {
                float v = acc4[t][r] + bn;
                int M = r + 8 * hi;
                hres[M * PX + t * 16 + n16] = v;
                hf[M * PH + t * 16 + n16]   = (_Float16)v;
            }
        }
    }

    // --- 14 transformer layers (seq_len=1: attn == out_proj(V)) ---
    for (int l = 0; l < NL; ++l) {
        const _Float16* Wv = ws + OFF_WV + l * 4096;
        const _Float16* Wo = ws + OFF_WO + l * 4096;
        const _Float16* W1 = ws + OFF_W1 + l * 8192;
        const _Float16* W2 = ws + OFF_W2 + l * 8192;
        const float* bv  = bqkv + l * 192 + 128;

        wmma_gemm<2, 4>(hf, PH, Wv, DM, lane, acc4);             // V = h @ Wv^T + bv
        store_h16<4, false>(acc4, sf, PS, bv, lane);

        wmma_gemm<2, 4>(sf, PS, Wo, DM, lane, acc4);             // attn = V @ Wo^T + bo
        store_f32<4>(acc4, x32, bo + l * DM, lane);

        ln_pass(hres, x32, hf, ln1w + l * DM, ln1b + l * DM, lane);  // h = LN(h+attn)

        wmma_gemm<2, 8>(hf, PH, W1, DM, lane, acc8);             // relu(h @ W1^T + b1)
        store_h16<8, true>(acc8, sf, PS, b1 + l * FF, lane);

        wmma_gemm<4, 4>(sf, PS, W2, FF, lane, acc4);             // ffn = . @ W2^T + b2
        store_f32<4>(acc4, x32, b2 + l * DM, lane);

        ln_pass(hres, x32, hf, ln2w + l * DM, ln2b + l * DM, lane);  // h = LN(h+ffn)
    }

    // --- feature head + policy MLP + action logits ---
    wmma_gemm<2, 8>(hf, PH, ws + OFF_WF, DM, lane, acc8);        // feats (no relu)
    store_h16<8, false>(acc8, sf, PS, bf, lane);

    wmma_gemm<4, 8>(sf, PS, ws + OFF_WP1, FF, lane, acc8);       // relu(p1)
    store_h16<8, true>(acc8, sf, PS, bp1, lane);

    wmma_gemm<4, 8>(sf, PS, ws + OFF_WP2, FF, lane, acc8);       // relu(p2)
    store_h16<8, true>(acc8, sf, PS, bp2, lane);

    wmma_gemm<4, 1>(sf, PS, ws + OFF_WA, FF, lane, acc4);        // logits [16x16], 7 valid
    {
        const int n = lane & 15;
        if (n < ACT) {
            float bn = ba[n];
            #pragma unroll
            for (int r = 0; r < 8; ++r) {
                int M = r + 8 * hi;
                out[(R0 + M) * ACT + n] = acc4[0][r] + bn;
            }
        }
    }
}

// ---------------- launch ----------------
extern "C" void kernel_launch(void* const* d_in, const int* in_sizes, int n_in,
                              void* d_out, int out_size, void* d_ws, size_t ws_size,
                              hipStream_t stream) {
    const float* board = (const float*)d_in[0];
    const int*   mark  = (const int*)d_in[1];
    const float* emb   = (const float*)d_in[2];
    const float* W_in  = (const float*)d_in[3];
    const float* b_in  = (const float*)d_in[4];
    const float* Wqkv  = (const float*)d_in[5];
    const float* bqkv  = (const float*)d_in[6];
    const float* Wo    = (const float*)d_in[7];
    const float* bo    = (const float*)d_in[8];
    const float* ln1w  = (const float*)d_in[9];
    const float* ln1b  = (const float*)d_in[10];
    const float* W1    = (const float*)d_in[11];
    const float* b1    = (const float*)d_in[12];
    const float* W2    = (const float*)d_in[13];
    const float* b2    = (const float*)d_in[14];
    const float* ln2w  = (const float*)d_in[15];
    const float* ln2b  = (const float*)d_in[16];
    const float* Wf    = (const float*)d_in[17];
    const float* bf    = (const float*)d_in[18];
    const float* Wp1   = (const float*)d_in[19];
    const float* bp1   = (const float*)d_in[20];
    const float* Wp2   = (const float*)d_in[21];
    const float* bp2   = (const float*)d_in[22];
    const float* Wa    = (const float*)d_in[23];
    const float* ba    = (const float*)d_in[24];
    _Float16* ws = (_Float16*)d_ws;
    float* out = (float*)d_out;

    // one-time-per-call weight conversion to f16 (tiny; L2-resident afterwards)
    k_cvt_win<<<16, 256, 0, stream>>>(W_in, ws + OFF_WIN);
    k_cvt_wv <<<224, 256, 0, stream>>>(Wqkv, ws + OFF_WV);
    k_cvt_plain<<<224, 256, 0, stream>>>(Wo,  ws + OFF_WO,  NL * 4096);
    k_cvt_plain<<<448, 256, 0, stream>>>(W1,  ws + OFF_W1,  NL * 8192);
    k_cvt_plain<<<448, 256, 0, stream>>>(W2,  ws + OFF_W2,  NL * 8192);
    k_cvt_plain<<<32, 256, 0, stream>>>(Wf,   ws + OFF_WF,  8192);
    k_cvt_plain<<<64, 256, 0, stream>>>(Wp1,  ws + OFF_WP1, 16384);
    k_cvt_plain<<<64, 256, 0, stream>>>(Wp2,  ws + OFF_WP2, 16384);
    k_cvt_wa <<<8, 256, 0, stream>>>(Wa, ws + OFF_WA);

    const int Bn = in_sizes[0] / BOARDN;   // 65536
    dim3 grid(Bn / 64), block(128);
    k_policy_forward<<<grid, block, 0, stream>>>(
        board, mark, emb, b_in, bqkv, bo, ln1w, ln1b, b1, b2, ln2w, ln2b,
        bf, bp1, bp2, ba, ws, out);
}
